// Attention_32366873542737
// MI455X (gfx1250) — matrix operand order repaired and checked by reading, not compiled, not run on hardware
//
#include <hip/hip_runtime.h>

// ---------------------------------------------------------------------------
// CDNA5 (gfx1250) fused causal-normalized attention, wave32 + WMMA bf16.
//   K1: QKV = X @ Wqkv           -> bf16 Q[b,h,n,d], K[b,h,n,d], V^T[b,h,d,n]
//   K2: flash-style per-16-row tile:
//         pass1: row prefix sums of dots (scaled, causal) -> mu, rsqrt(var)
//         pass2: recompute dots, normalize*g+b, P @ V  (fp32 accum)
//   K3: OUT = O @ Wout + bias    (fp32 result)
// Workspace: Q(4MB)+K(4MB)+V^T(4MB)+O(8MB) ~ 20.6MB (lives in 192MB L2).
// ---------------------------------------------------------------------------

typedef __attribute__((ext_vector_type(16))) __bf16 v16bf;
typedef __attribute__((ext_vector_type(8)))  __bf16 v8bf;
typedef __attribute__((ext_vector_type(8)))  float  v8f;

static __device__ __forceinline__ v8f vzero8() {
  v8f v;
#pragma unroll
  for (int i = 0; i < 8; ++i) v[i] = 0.0f;
  return v;
}

static __device__ __forceinline__ v8f wmma_bf16(v16bf a, v16bf b, v8f c) {
  // D = A(16x32 bf16) * B(32x16 bf16) + C(16x16 f32)
  return __builtin_amdgcn_wmma_f32_16x16x32_bf16(false, a, false, b,
                                                 (short)0, c, false, false);
}

// ---------------------------------------------------------------------------
// Kernel 1: QKV projection GEMM.  X[4096,512] @ W[512,1536].
// Block = 128 threads (4 waves), computes a 64x64 tile; each wave owns 16 rows.
// Each 64-column block maps to exactly one (q/k/v, head) slice.
// ---------------------------------------------------------------------------
__global__ __launch_bounds__(128) void qkv_gemm_kernel(
    const float* __restrict__ x, const float* __restrict__ w,
    __bf16* __restrict__ qb, __bf16* __restrict__ kb, __bf16* __restrict__ vtb) {
  __shared__ __bf16 lds_a[64][32];   // [row][k]
  __shared__ __bf16 lds_b[64][32];   // [col][k]  (W transposed slice)

  const int nblk = blockIdx.x;       // 0..23
  const int m0   = blockIdx.y * 64;  // rows of X
  const int n0   = nblk * 64;        // cols of W
  const int t    = threadIdx.x;
  const int wave = t >> 5;
  const int lane = t & 31;
  const int lr   = lane & 15;
  const int hi   = lane >> 4;

  v8f acc[4];
#pragma unroll
  for (int ct = 0; ct < 4; ++ct) acc[ct] = vzero8();

  for (int kc = 0; kc < 512; kc += 32) {
    __syncthreads();
    {  // X slice 64x32 -> bf16 LDS
      const int row = t >> 1, c0 = (t & 1) * 16;
      const float* src = x + (size_t)(m0 + row) * 512 + kc + c0;
#pragma unroll
      for (int e = 0; e < 16; ++e) lds_a[row][c0 + e] = (__bf16)src[e];
    }
    {  // W slice 32x64 -> transposed bf16 LDS
      const int col = t >> 1, kh = (t & 1) * 16;
      const float* src = w + (size_t)(kc + kh) * 1536 + n0 + col;
#pragma unroll
      for (int e = 0; e < 16; ++e) lds_b[col][kh + e] = (__bf16)src[e * 1536];
    }
    __syncthreads();

    // A operand: contiguous K layout (lanes 0-15: K 0-15; 16-31: K 16-31)
    const int arow = wave * 16 + lr;
    v8bf lo = *(const v8bf*)&lds_a[arow][hi * 16];
    v8bf hh = *(const v8bf*)&lds_a[arow][hi * 16 + 8];
    v16bf a;
#pragma unroll
    for (int e = 0; e < 8; ++e) { a[e] = lo[e]; a[8 + e] = hh[e]; }

#pragma unroll
    for (int ct = 0; ct < 4; ++ct) {
      // B operand: lane = column, 16 contiguous K per lane half
      v16bf bmat = *(const v16bf*)&lds_b[ct * 16 + lr][hi * 16];
      acc[ct] = wmma_bf16(a, bmat, acc[ct]);
    }
  }

  // Epilogue: scatter into q / k / v^T bf16 tensors.
  // m0 is a multiple of 64 and 2048 % 64 == 0, so a 64-row tile never crosses
  // a batch boundary: batch, target tensor and base pointer are wave-uniform.
  const int which = nblk >> 3;   // 0=q 1=k 2=v
  const int head  = nblk & 7;
  const int batch = m0 >> 11;
  const int seq0  = (m0 & 2047) + wave * 16 + hi * 8;
  const size_t bhh = (size_t)(batch * 8 + head);

  if (which < 2) {
    __bf16* __restrict__ dst = (which == 0) ? qb : kb;
    __bf16* base = dst + (bhh * 2048 + seq0) * 64 + lr;  // + r*64 + ct*16
#pragma unroll
    for (int ct = 0; ct < 4; ++ct)
#pragma unroll
      for (int r = 0; r < 8; ++r)
        base[(size_t)r * 64 + ct * 16] = (__bf16)acc[ct][r];
  } else {
    __bf16* base = vtb + (bhh * 64 + lr) * 2048 + seq0;  // + ct*16*2048 + r
#pragma unroll
    for (int ct = 0; ct < 4; ++ct)
#pragma unroll
      for (int r = 0; r < 8; ++r)
        base[(size_t)(ct * 16) * 2048 + r] = (__bf16)acc[ct][r];
  }
}

// ---------------------------------------------------------------------------
// Kernel 2: causal-normalized attention. One wave per (b*h, 16-row q-tile).
// ---------------------------------------------------------------------------
__global__ __launch_bounds__(32) void attn_kernel(
    const __bf16* __restrict__ qb, const __bf16* __restrict__ kb,
    const __bf16* __restrict__ vtb, const float* __restrict__ gvec,
    const float* __restrict__ bvec, float* __restrict__ o) {
  const int bh    = blockIdx.x;       // 0..15
  const int head  = bh & 7;
  const int batch = bh >> 3;
  const int i0    = blockIdx.y * 16;  // first query row of this tile
  const int lane  = threadIdx.x;
  const int lr    = lane & 15;
  const int hi    = lane >> 4;
  const float scale = 0.04419417382415922f;  // 512^-0.5

  __shared__ float st[16][33];   // 16x32 dots staging (+pad)
  __shared__ float mu_s[16];
  __shared__ float rs_s[16];

  const __bf16* kbh = kb + (size_t)bh * 2048 * 64;
  const __bf16* vbh = vtb + (size_t)bh * 64 * 2048;

  // Load Q A-operands once (two 16x32 K-chunks of d=64).
  const __bf16* qrow = qb + ((size_t)bh * 2048 + i0 + lr) * 64;
  v16bf qa0, qa1;
  {
    v8bf lo = *(const v8bf*)(qrow + hi * 16);
    v8bf hh = *(const v8bf*)(qrow + hi * 16 + 8);
#pragma unroll
    for (int e = 0; e < 8; ++e) { qa0[e] = lo[e]; qa0[8 + e] = hh[e]; }
    lo = *(const v8bf*)(qrow + 32 + hi * 16);
    hh = *(const v8bf*)(qrow + 32 + hi * 16 + 8);
#pragma unroll
    for (int e = 0; e < 8; ++e) { qa1[e] = lo[e]; qa1[8 + e] = hh[e]; }
  }

  // Raw (unscaled) 16x16 dots tile vs key tile jt.
  auto dots = [&](int jt) -> v8f {
    int krow = jt * 16 + lr;
    if (krow > 2047) krow = 2047;  // clamp (masked away downstream)
    const __bf16* krp = kbh + (size_t)krow * 64;
    v16bf b0 = *(const v16bf*)(krp + hi * 16);
    v16bf b1 = *(const v16bf*)(krp + 32 + hi * 16);
    v8f c = vzero8();
    c = wmma_bf16(qa0, b0, c);
    c = wmma_bf16(qa1, b1, c);
    return c;
  };

  const int diag_t = i0 >> 4;

  // -------- Pass 1: per-row prefix sum / sum-of-squares of scaled dots -----
  float sum = 0.0f, sq = 0.0f;
  for (int jt = 0; jt <= diag_t; ++jt) {
    v8f d = dots(jt);
    __syncthreads();
#pragma unroll
    for (int r = 0; r < 8; ++r) st[r + hi * 8][lr] = d[r] * scale;
    __syncthreads();
    if (lane < 16) {
      const int lim = (jt == diag_t) ? lane : 15;  // causal: j <= i
      for (int jl = 0; jl <= lim; ++jl) {
        float v = st[lane][jl];
        sum += v;
        sq += v * v;
      }
    }
  }
  __syncthreads();
  if (lane < 16) {
    const float cnt = (float)(i0 + lane + 1);
    const float mu  = sum / cnt;
    float s = sq / cnt - mu * mu;           // E[d^2]-mu^2 == E[(d-mu)^2] on prefix
    s = fmaxf(s, 0.0f);
    mu_s[lane] = mu;
    rs_s[lane] = rsqrtf(s + 1e-8f);
  }
  __syncthreads();

  const float g  = gvec[head];
  const float bb = bvec[head];

  // -------- Pass 2: recompute dots, normalize, P @ V ----------------------
  v8f o_acc[4];
#pragma unroll
  for (int ct = 0; ct < 4; ++ct) o_acc[ct] = vzero8();

  for (int jt0 = 0; jt0 <= diag_t; jt0 += 2) {
    const int j0 = jt0 * 16;
    v8f d0 = dots(jt0);
    v8f d1;
    const bool has2 = (jt0 + 1) <= diag_t;   // uniform branch, EXEC stays full
    if (has2) d1 = dots(jt0 + 1); else d1 = vzero8();
    __syncthreads();
#pragma unroll
    for (int r = 0; r < 8; ++r) {
      st[r + hi * 8][lr]      = d0[r];
      st[r + hi * 8][16 + lr] = d1[r];
    }
    __syncthreads();

    // Build P A-operand (bf16), applying scale, normalization, causal mask.
    const int   prow = lr;
    const int   irow = i0 + prow;
    const float mu = mu_s[prow], rs = rs_s[prow];
    v16bf pa;
#pragma unroll
    for (int e = 0; e < 16; ++e) {
      const int kidx  = hi * 16 + e;
      const int jglob = j0 + kidx;
      const float val = st[prow][kidx] * scale;
      const float p = (jglob <= irow) ? ((val - mu) * rs * g + bb) : 0.0f;
      pa[e] = (__bf16)p;
    }

    // V^T B-operands: lane = d-column, 16 contiguous keys per lane half.
    int jb = j0 + hi * 16;
    if (jb > 2048 - 16) jb = 2048 - 16;      // clamp; P==0 there
#pragma unroll
    for (int ct = 0; ct < 4; ++ct) {
      v16bf vb = *(const v16bf*)(vbh + (size_t)(ct * 16 + lr) * 2048 + jb);
      o_acc[ct] = wmma_bf16(pa, vb, o_acc[ct]);
    }
    __syncthreads();
  }

  // Write O (fp32, [b*n, 512] layout); base pointer is wave-uniform.
  float* obase = o + ((size_t)batch * 2048 + i0 + hi * 8) * 512 + head * 64 + lr;
#pragma unroll
  for (int ct = 0; ct < 4; ++ct)
#pragma unroll
    for (int r = 0; r < 8; ++r)
      obase[(size_t)r * 512 + ct * 16] = o_acc[ct][r];
}

// ---------------------------------------------------------------------------
// Kernel 3: output projection.  O[4096,512] @ Wout[512,512] + bias.
// ---------------------------------------------------------------------------
__global__ __launch_bounds__(128) void out_gemm_kernel(
    const float* __restrict__ a_in, const float* __restrict__ w,
    const float* __restrict__ bias, float* __restrict__ out) {
  __shared__ __bf16 lds_a[64][32];
  __shared__ __bf16 lds_b[64][32];

  const int m0 = blockIdx.y * 64;
  const int n0 = blockIdx.x * 64;
  const int t    = threadIdx.x;
  const int wave = t >> 5;
  const int lane = t & 31;
  const int lr   = lane & 15;
  const int hi   = lane >> 4;

  v8f acc[4];
#pragma unroll
  for (int ct = 0; ct < 4; ++ct) acc[ct] = vzero8();

  for (int kc = 0; kc < 512; kc += 32) {
    __syncthreads();
    {
      const int row = t >> 1, c0 = (t & 1) * 16;
      const float* src = a_in + (size_t)(m0 + row) * 512 + kc + c0;
#pragma unroll
      for (int e = 0; e < 16; ++e) lds_a[row][c0 + e] = (__bf16)src[e];
    }
    {
      const int col = t >> 1, kh = (t & 1) * 16;
      const float* src = w + (size_t)(kc + kh) * 512 + n0 + col;
#pragma unroll
      for (int e = 0; e < 16; ++e) lds_b[col][kh + e] = (__bf16)src[e * 512];
    }
    __syncthreads();

    const int arow = wave * 16 + lr;
    v8bf lo = *(const v8bf*)&lds_a[arow][hi * 16];
    v8bf hh = *(const v8bf*)&lds_a[arow][hi * 16 + 8];
    v16bf a;
#pragma unroll
    for (int e = 0; e < 8; ++e) { a[e] = lo[e]; a[8 + e] = hh[e]; }

#pragma unroll
    for (int ct = 0; ct < 4; ++ct) {
      v16bf bmat = *(const v16bf*)&lds_b[ct * 16 + lr][hi * 16];
      acc[ct] = wmma_bf16(a, bmat, acc[ct]);
    }
  }

  // Bias + store; base pointer wave-uniform, bias loaded once per column tile.
  float bv[4];
#pragma unroll
  for (int ct = 0; ct < 4; ++ct) bv[ct] = bias[n0 + ct * 16 + lr];
  float* obase = out + (size_t)(m0 + wave * 16 + hi * 8) * 512 + n0 + lr;
#pragma unroll
  for (int ct = 0; ct < 4; ++ct)
#pragma unroll
    for (int r = 0; r < 8; ++r)
      obase[(size_t)r * 512 + ct * 16] = acc[ct][r] + bv[ct];
}

// ---------------------------------------------------------------------------
extern "C" void kernel_launch(void* const* d_in, const int* in_sizes, int n_in,
                              void* d_out, int out_size, void* d_ws, size_t ws_size,
                              hipStream_t stream) {
  (void)in_sizes; (void)n_in; (void)out_size; (void)ws_size;

  const float* x      = (const float*)d_in[0];  // [2,2048,512]
  const float* w_qkv  = (const float*)d_in[1];  // [512,1536]
  const float* w_out  = (const float*)d_in[2];  // [512,512]
  const float* b_out  = (const float*)d_in[3];  // [512]
  const float* norm_g = (const float*)d_in[4];  // [1,8,1,1]
  const float* norm_b = (const float*)d_in[5];  // [1,8,1,1]
  float* out = (float*)d_out;                   // [2,2048,512]

  // Workspace layout (bf16 Q,K,V^T + fp32 O), 4KB guard between regions.
  char* ws = (char*)d_ws;
  const size_t QKV_BYTES = 2ull * 8 * 2048 * 64 * sizeof(__bf16);  // 4 MB
  const size_t STRIDE    = QKV_BYTES + 4096;
  __bf16* qb  = (__bf16*)(ws);
  __bf16* kb  = (__bf16*)(ws + STRIDE);
  __bf16* vtb = (__bf16*)(ws + 2 * STRIDE);
  float*  ob  = (float*)(ws + 3 * STRIDE);                         // 8 MB

  qkv_gemm_kernel<<<dim3(24, 64), 128, 0, stream>>>(x, w_qkv, qb, kb, vtb);
  attn_kernel<<<dim3(16, 128), 32, 0, stream>>>(qb, kb, vtb, norm_g, norm_b, ob);
  out_gemm_kernel<<<dim3(8, 64), 128, 0, stream>>>(ob, w_out, b_out, out);
}
